// SlidingWindowAttention_89644557402799
// MI455X (gfx1250) — compile-verified, matrix operand order checked by reading, not run on hardware
//
#include <hip/hip_runtime.h>
#include <hip/hip_bf16.h>

#define B_      2
#define T_      2048
#define DM      1024
#define H_      16
#define DH_     64
#define HDH     1024
#define WINDOW_ 512

typedef __attribute__((ext_vector_type(16))) __bf16 v16bf;
typedef __attribute__((ext_vector_type(8)))  __bf16 v8bf;
typedef __attribute__((ext_vector_type(8)))  float  v8f;

union FragBF {
    v16bf v;
    v8bf  h[2];
};

static __device__ __forceinline__ v8f wmma_bf16(const FragBF& a, const FragBF& b, v8f c) {
    // 8 args: (neg_a, A, neg_b, B, c_mod, C, reuse_a, reuse_b)
    return __builtin_amdgcn_wmma_f32_16x16x32_bf16(false, a.v, false, b.v, (short)0, c, false, false);
}

// ---------------------------------------------------------------- async global->LDS
#if defined(__gfx1250__)
#define ASYNC_LDS 1
#else
#define ASYNC_LDS 0
#endif

// copy 16 bytes (8 bf16) from global to LDS, per-lane, via GLOBAL_LOAD_ASYNC_TO_LDS_B128
// (GV mode: VDST = LDS byte offset VGPR, VADDR = 64-bit global address VGPR pair)
static __device__ __forceinline__ void cp16_g2l(__bf16* l, const __bf16* g) {
#if ASYNC_LDS
    unsigned int lds_off = (unsigned int)(unsigned long long)l;  // low 32 bits = LDS byte addr
    asm volatile("global_load_async_to_lds_b128 %0, %1, off"
                 :
                 : "v"(lds_off), "v"(g)
                 : "memory");
#else
    *(v8bf*)l = *(const v8bf*)g;
#endif
}

static __device__ __forceinline__ void wait_async_lds() {
#if ASYNC_LDS
#if __has_builtin(__builtin_amdgcn_s_wait_asynccnt)
    __builtin_amdgcn_s_wait_asynccnt(0);
#else
    asm volatile("s_wait_asynccnt 0" ::: "memory");
#endif
#endif
}

// ---------------------------------------------------------------- converts
__global__ void cvt_f32_bf16(const float* __restrict__ in, __bf16* __restrict__ out, int n) {
    int i = blockIdx.x * blockDim.x + threadIdx.x;
    int stride = gridDim.x * blockDim.x;
    for (; i < n; i += stride) out[i] = (__bf16)in[i];
}

// w: (K=1024, N=1024) row-major -> wt: (N, K) row-major bf16
__global__ void transpose_cvt_1024(const float* __restrict__ w, __bf16* __restrict__ wt) {
    int idx = blockIdx.x * blockDim.x + threadIdx.x;   // idx = n*1024 + k  (exactly 1<<20 threads)
    int n = idx >> 10, k = idx & 1023;
    wt[idx] = (__bf16)w[(k << 10) + n];
}

// v: (B*T, H*DH) bf16 -> vt: (B*H*DH, T) bf16
__global__ void v_transpose(const __bf16* __restrict__ v, __bf16* __restrict__ vt) {
    int idx = blockIdx.x * blockDim.x + threadIdx.x;   // ((b*H+h)*DH+d)*T + t
    int t = idx & (T_ - 1);
    int d = (idx >> 11) & (DH_ - 1);
    int h = (idx >> 17) & (H_ - 1);
    int b = idx >> 21;
    vt[idx] = v[((size_t)(b * T_ + t) << 10) + h * DH_ + d];
}

// ---------------------------------------------------------------- GEMM
// C(M,N) = A(M,K) * Bt(N,K)^T, bf16 in, f32 accumulate, OutT out.
// Double-buffered LDS, async global->LDS staging, one barrier per K-step.
template <typename OutT>
__global__ __launch_bounds__(256)
void gemm_bf16_wmma(const __bf16* __restrict__ A,
                    const __bf16* __restrict__ Bt,
                    OutT* __restrict__ C,
                    int M, int N, int K) {
    constexpr int BM = 128, BN = 128, BK = 32, LDT = BK + 8;
    __shared__ __align__(16) __bf16 As[2][BM][LDT];
    __shared__ __align__(16) __bf16 Bs[2][BN][LDT];

    const int tid  = threadIdx.x;
    const int lane = tid & 31;
    const int wid  = tid >> 5;       // 0..7
    const int wm   = wid & 1;        // 2 waves along M
    const int wn   = wid >> 1;       // 4 waves along N
    const int m0   = blockIdx.y * BM;
    const int n0   = blockIdx.x * BN;
    const int hlf  = lane >> 4;
    const int l15  = lane & 15;
    const int kb   = hlf * 8;        // 16-bit operand K sub-offset

    // each thread copies two 16B chunks per tile (512 chunks, 256 threads)
    const int r0 = tid >> 2;               // 0..63
    const int c0 = (tid & 3) << 3;         // 0,8,16,24

    auto load_tiles = [&](int k0, int buf) {
        cp16_g2l(&As[buf][r0][c0],      &A [(size_t)(m0 + r0)      * K + k0 + c0]);
        cp16_g2l(&As[buf][r0 + 64][c0], &A [(size_t)(m0 + r0 + 64) * K + k0 + c0]);
        cp16_g2l(&Bs[buf][r0][c0],      &Bt[(size_t)(n0 + r0)      * K + k0 + c0]);
        cp16_g2l(&Bs[buf][r0 + 64][c0], &Bt[(size_t)(n0 + r0 + 64) * K + k0 + c0]);
    };

    const v8f vzero = {};
    v8f acc[4][2];
    #pragma unroll
    for (int i = 0; i < 4; ++i)
        #pragma unroll
        for (int j = 0; j < 2; ++j) acc[i][j] = vzero;

    load_tiles(0, 0);
    wait_async_lds();
    __syncthreads();

    for (int k0 = 0; k0 < K; k0 += BK) {
        const int cur = (k0 / BK) & 1;
        if (k0 + BK < K) load_tiles(k0 + BK, cur ^ 1);
        if (k0 + 2 * BK < K) {
            __builtin_prefetch(&A [(size_t)(m0 + r0) * K + k0 + 2 * BK + c0], 0, 0);
            __builtin_prefetch(&Bt[(size_t)(n0 + r0) * K + k0 + 2 * BK + c0], 0, 0);
        }

        FragBF af[4], bfr[2];
        #pragma unroll
        for (int i = 0; i < 4; ++i) {
            int row = wm * 64 + i * 16 + l15;
            af[i].h[0] = *(const v8bf*)&As[cur][row][kb];
            af[i].h[1] = *(const v8bf*)&As[cur][row][kb + 16];
        }
        #pragma unroll
        for (int j = 0; j < 2; ++j) {
            int row = wn * 32 + j * 16 + l15;
            bfr[j].h[0] = *(const v8bf*)&Bs[cur][row][kb];
            bfr[j].h[1] = *(const v8bf*)&Bs[cur][row][kb + 16];
        }
        #pragma unroll
        for (int i = 0; i < 4; ++i)
            #pragma unroll
            for (int j = 0; j < 2; ++j)
                acc[i][j] = wmma_bf16(af[i], bfr[j], acc[i][j]);

        wait_async_lds();
        __syncthreads();
    }

    #pragma unroll
    for (int i = 0; i < 4; ++i)
        #pragma unroll
        for (int j = 0; j < 2; ++j) {
            int n = n0 + wn * 32 + j * 16 + l15;
            #pragma unroll
            for (int r = 0; r < 8; ++r) {
                int m = m0 + wm * 64 + i * 16 + hlf * 8 + r;
                C[(size_t)m * N + n] = (OutT)acc[i][j][r];
            }
        }
}

// ---------------------------------------------------------------- attention
// Q,K: (B*T, HDH) bf16; Vt: (B*H*DH, T) bf16; O: (B*T, HDH) bf16
__global__ __launch_bounds__(128)
void swa_attention(const __bf16* __restrict__ Q,
                   const __bf16* __restrict__ Kb,
                   const __bf16* __restrict__ Vt,
                   __bf16* __restrict__ O) {
    __shared__ __align__(16) __bf16 Ks[32][72];     // 32 keys x 64 dh (+pad)
    __shared__ __align__(16) __bf16 Vts[64][40];    // 64 dh x 32 keys (+pad)
    __shared__ __align__(16) __bf16 Ps[4][16][40];  // per-wave P bounce buffer

    const int tid  = threadIdx.x;
    const int lane = tid & 31;
    const int wid  = tid >> 5;   // 0..3
    const int l15  = lane & 15;
    const int hlf  = lane >> 4;
    const int kb   = hlf * 8;

    const int b    = blockIdx.z;
    const int h    = blockIdx.y;
    const int q0wg = blockIdx.x * 64;
    const int q0   = q0wg + wid * 16;

    // staging slots for this thread
    const int kr = tid >> 3, kc8 = (tid & 7) << 3;   // K chunk: rows 0..15 (+16)
    const int vr = tid >> 2, vc8 = (tid & 3) << 3;   // Vt chunk: rows 0..31 (+32)

    // Q fragments (16 rows x 64 dh) in registers for the whole key loop
    FragBF qf[2];
    {
        const __bf16* qrow = Q + (size_t)(b * T_ + q0 + l15) * HDH + h * DH_;
        #pragma unroll
        for (int f = 0; f < 2; ++f) {
            qf[f].h[0] = *(const v8bf*)&qrow[f * 32 + kb];
            qf[f].h[1] = *(const v8bf*)&qrow[f * 32 + kb + 16];
        }
    }

    const v8f vzero = {};
    v8f accO[4] = {vzero, vzero, vzero, vzero};
    float mi[8], li[8];
    #pragma unroll
    for (int r = 0; r < 8; ++r) { mi[r] = -1e30f; li[r] = 0.0f; }

    int kstart = q0wg - WINDOW_;
    if (kstart < 0) kstart = 0;
    const int kend = q0wg + 64;

    for (int kc = kstart; kc < kend; kc += 32) {
        __syncthreads();
        // stage K chunk (32x64) and Vt chunk (64x32) via async copies
        cp16_g2l(&Ks[kr][kc8],
                 &Kb[(size_t)(b * T_ + kc + kr) * HDH + h * DH_ + kc8]);
        cp16_g2l(&Ks[kr + 16][kc8],
                 &Kb[(size_t)(b * T_ + kc + kr + 16) * HDH + h * DH_ + kc8]);
        cp16_g2l(&Vts[vr][vc8],
                 &Vt[(size_t)((b * H_ + h) * DH_ + vr) * T_ + kc + vc8]);
        cp16_g2l(&Vts[vr + 32][vc8],
                 &Vt[(size_t)((b * H_ + h) * DH_ + vr + 32) * T_ + kc + vc8]);
        if (kc + 32 < kend) {
            __builtin_prefetch(&Kb[(size_t)(b * T_ + kc + 32 + kr) * HDH + h * DH_ + kc8], 0, 0);
            __builtin_prefetch(&Vt[(size_t)((b * H_ + h) * DH_ + vr) * T_ + kc + 32 + vc8], 0, 0);
        }
        wait_async_lds();
        __syncthreads();

        // scores: two 16-key sub-tiles, each = 2 WMMAs over DH=64
        v8f s[2];
        #pragma unroll
        for (int st = 0; st < 2; ++st) {
            FragBF kf[2];
            int krow = st * 16 + l15;
            #pragma unroll
            for (int f = 0; f < 2; ++f) {
                kf[f].h[0] = *(const v8bf*)&Ks[krow][f * 32 + kb];
                kf[f].h[1] = *(const v8bf*)&Ks[krow][f * 32 + kb + 16];
            }
            v8f t = wmma_bf16(qf[0], kf[0], vzero);
            t     = wmma_bf16(qf[1], kf[1], t);
            int key = kc + st * 16 + l15;
            #pragma unroll
            for (int r = 0; r < 8; ++r) {
                int qi = q0 + hlf * 8 + r;
                float sv = t[r] * 0.125f;                 // 1/sqrt(64)
                bool ok = (key <= qi) && (qi - key <= WINDOW_);
                t[r] = ok ? sv : -1e30f;
            }
            s[st] = t;
        }

        // online softmax (each row spread across 16 lanes of a half-wave)
        #pragma unroll
        for (int r = 0; r < 8; ++r) {
            float rmax = fmaxf(s[0][r], s[1][r]);
            #pragma unroll
            for (int off = 1; off < 16; off <<= 1)
                rmax = fmaxf(rmax, __shfl_xor(rmax, off, 32));
            float mnew  = fmaxf(mi[r], rmax);
            float alpha = __expf(mi[r] - mnew);
            mi[r] = mnew;
            float p0 = __expf(s[0][r] - mnew);
            float p1 = __expf(s[1][r] - mnew);
            s[0][r] = p0; s[1][r] = p1;
            float rsum = p0 + p1;
            #pragma unroll
            for (int off = 1; off < 16; off <<= 1)
                rsum += __shfl_xor(rsum, off, 32);
            li[r] = li[r] * alpha + rsum;
            #pragma unroll
            for (int j = 0; j < 4; ++j) accO[j][r] *= alpha;
        }

        // bounce P through LDS: C-layout -> A-operand layout
        #pragma unroll
        for (int st = 0; st < 2; ++st)
            #pragma unroll
            for (int r = 0; r < 8; ++r)
                Ps[wid][hlf * 8 + r][st * 16 + l15] = (__bf16)s[st][r];

        FragBF pf;
        pf.h[0] = *(const v8bf*)&Ps[wid][l15][kb];
        pf.h[1] = *(const v8bf*)&Ps[wid][l15][kb + 16];

        #pragma unroll
        for (int j = 0; j < 4; ++j) {
            FragBF vf;
            int vrow = j * 16 + l15;
            vf.h[0] = *(const v8bf*)&Vts[vrow][kb];
            vf.h[1] = *(const v8bf*)&Vts[vrow][kb + 16];
            accO[j] = wmma_bf16(pf, vf, accO[j]);
        }
    }

    #pragma unroll
    for (int j = 0; j < 4; ++j) {
        int dh = j * 16 + l15;
        #pragma unroll
        for (int r = 0; r < 8; ++r) {
            int qi = q0 + hlf * 8 + r;
            float ov = accO[j][r] / li[r];
            O[(size_t)(b * T_ + qi) * HDH + h * DH_ + dh] = (__bf16)ov;
        }
    }
}

// ---------------------------------------------------------------- launch
extern "C" void kernel_launch(void* const* d_in, const int* in_sizes, int n_in,
                              void* d_out, int out_size, void* d_ws, size_t ws_size,
                              hipStream_t stream) {
    (void)in_sizes; (void)n_in; (void)out_size; (void)ws_size;
    const float* x  = (const float*)d_in[0];
    const float* wq = (const float*)d_in[1];
    const float* wk = (const float*)d_in[2];
    const float* wv = (const float*)d_in[3];
    const float* wo = (const float*)d_in[4];
    float* out = (float*)d_out;

    const size_t SZ_X = (size_t)B_ * T_ * DM;   // 4M elements
    const size_t SZ_W = (size_t)DM * HDH;       // 1M elements

    __bf16* base = (__bf16*)d_ws;
    __bf16* xb   = base;
    __bf16* wqt  = xb + SZ_X;
    __bf16* wkt  = wqt + SZ_W;
    __bf16* wvt  = wkt + SZ_W;
    __bf16* wot  = wvt + SZ_W;
    __bf16* qb   = wot + SZ_W;
    __bf16* kbuf = qb + SZ_X;
    __bf16* vbuf = kbuf + SZ_X;
    __bf16* vtb  = vbuf + SZ_X;
    __bf16* attn = vtb + SZ_X;

    const int M = B_ * T_;   // 4096

    cvt_f32_bf16<<<4096, 256, 0, stream>>>(x, xb, (int)SZ_X);
    transpose_cvt_1024<<<4096, 256, 0, stream>>>(wq, wqt);
    transpose_cvt_1024<<<4096, 256, 0, stream>>>(wk, wkt);
    transpose_cvt_1024<<<4096, 256, 0, stream>>>(wv, wvt);
    transpose_cvt_1024<<<4096, 256, 0, stream>>>(wo, wot);

    dim3 gg(HDH / 128, M / 128);   // (8, 32)
    gemm_bf16_wmma<__bf16><<<gg, 256, 0, stream>>>(xb, wqt, qb,   M, HDH, DM);
    gemm_bf16_wmma<__bf16><<<gg, 256, 0, stream>>>(xb, wkt, kbuf, M, HDH, DM);
    gemm_bf16_wmma<__bf16><<<gg, 256, 0, stream>>>(xb, wvt, vbuf, M, HDH, DM);

    v_transpose<<<16384, 256, 0, stream>>>(vbuf, vtb);

    dim3 ga(T_ / 64, H_, B_);      // (32, 16, 2)
    swa_attention<<<ga, 128, 0, stream>>>(qb, kbuf, vtb, attn);

    dim3 go(DM / 128, M / 128);
    gemm_bf16_wmma<float><<<go, 256, 0, stream>>>(attn, wot, out, M, DM, HDH);
}
